// Encoder_Decoder_Attention_90091234001524
// MI455X (gfx1250) — matrix operand order repaired and checked by reading, not compiled, not run on hardware
//
#include <hip/hip_runtime.h>
#include <cstddef>

// ---------------------------------------------------------------------------
// Encoder-decoder attention, MI455X (gfx1250), wave32 + WMMA bf16.
//   B=8, Sq=Sk=1024, HIDDEN=1024, N_HEAD=16, D_HEAD=64
// All matmuls run on v_wmma_f32_16x16x32_bf16 with f32 accumulators.
// ---------------------------------------------------------------------------

#define BB 8
#define SS 1024
#define HH 1024
#define NHEAD 16
#define DHEAD 64

typedef __attribute__((ext_vector_type(16))) unsigned short v16u;
typedef __attribute__((ext_vector_type(16))) __bf16         v16bf;
typedef __attribute__((ext_vector_type(8)))  float          v8f;
typedef __attribute__((ext_vector_type(4)))  unsigned int   v4u;
// Matches the async-copy builtin's pointee type exactly (GCC-style vector).
typedef int v4i_n __attribute__((vector_size(16)));

#define AS_GLOBAL __attribute__((address_space(1)))
#define AS_LDS    __attribute__((address_space(3)))

// Async global->LDS copy (CDNA5 ASYNCcnt path) if the toolchain declares it.
#if __has_builtin(__builtin_amdgcn_global_load_async_to_lds_b128) && \
    __has_builtin(__builtin_amdgcn_s_wait_asynccnt)
#define HAVE_ASYNC_COPY 1
#else
#define HAVE_ASYNC_COPY 0
#endif

// f32 -> bf16, round-to-nearest-even
static __device__ __forceinline__ unsigned short f2bf(float f) {
  unsigned int u = __builtin_bit_cast(unsigned int, f);
  u += 0x7fffu + ((u >> 16) & 1u);
  return (unsigned short)(u >> 16);
}

static __device__ __forceinline__ v8f wmma_bf16(v16u a, v16u b, v8f c) {
  return __builtin_amdgcn_wmma_f32_16x16x32_bf16(
      false, __builtin_bit_cast(v16bf, a),
      false, __builtin_bit_cast(v16bf, b),
      (short)0, c, false, false);
}

// A fragment (16x32, M x K), row-major source with leading dim `ld` (elements).
// Layout: lanes 0-15 -> rows 0-15, h[0..15]=K0..15;
//         lanes 16-31 -> rows 0-15, h[0..15]=K16..31.
// => per lane: two contiguous 16B chunks at K offsets k0 and k0+8.
static __device__ __forceinline__ v16u load_a_frag(const unsigned short* base, int ld) {
  int lane = threadIdx.x & 31;
  const unsigned short* p = base + (size_t)(lane & 15) * ld + ((lane >> 4) << 4);
  union { v16u v; v4u q[2]; } u;
  u.q[0] = *(const v4u*)(p);
  u.q[1] = *(const v4u*)(p + 8);
  return u.v;
}

// B fragment (32x16, K x N) where B = Wt^T and Wt is row-major [N][K]:
// this code's B order (same as A) uses h[m] = B[k0+m][col] = Wt[col][k0+m] -> 16 contiguous
// elements of one Wt row: a single 32B load per lane.
static __device__ __forceinline__ v16u load_bT_frag(const unsigned short* base, int ld) {
  int lane = threadIdx.x & 31;
  const unsigned short* p = base + (size_t)(lane & 15) * ld + ((lane >> 4) << 4);
  return *(const v16u*)p;
}

// B fragment (32x16, K x N) from a row-major K x N buffer in LDS (gather).
static __device__ __forceinline__ v16u load_b_frag_lds(const unsigned short* base, int ld) {
  int lane = threadIdx.x & 31;
  int col  = lane & 15;
  int k0   = (lane >> 4) << 4;
  v16u r;
#pragma unroll
  for (int j = 0; j < 8; ++j) {
    r[2 * j]     = base[(k0 + 2 * j)     * ld + col];
    r[2 * j + 1] = base[(k0 + 2 * j + 1) * ld + col];
  }
  return r;
}

// ---------------------------------------------------------------------------
__global__ void cvt_f32_bf16(const float* __restrict__ src,
                             unsigned short* __restrict__ dst, int n) {
  int i = blockIdx.x * blockDim.x + threadIdx.x;
  if (i < n) dst[i] = f2bf(src[i]);
}

// ---------------------------------------------------------------------------
// Y[M,N] = A[M,K](bf16) * W[N,K]^T (bf16). 256 thr = 8 waves; block tile
// 256x64; each wave: 32 rows x 64 cols = 8 C tiles. Per 32-K step: 2 A-frags
// + 4 B-frags -> 8 WMMA (each B frag reused across both A frags).
// FINAL: add bias, write f32; else write bf16.
template <bool FINAL>
__global__ __launch_bounds__(256) void gemm_xwT(
    const unsigned short* __restrict__ A, const unsigned short* __restrict__ W,
    unsigned short* __restrict__ Ybf, float* __restrict__ Yf,
    const float* __restrict__ bias, int M, int N, int K) {
  int wave = threadIdx.x >> 5;
  int lane = threadIdx.x & 31;
  int m0 = blockIdx.x * 256 + wave * 32;
  int n0 = blockIdx.y * 64;
  v8f acc[2][4] = {};
  const unsigned short* A0 = A + (size_t)m0 * K;
  const unsigned short* A1 = A0 + (size_t)16 * K;
  for (int k0 = 0; k0 < K; k0 += 32) {
    v16u a0 = load_a_frag(A0 + k0, K);
    v16u a1 = load_a_frag(A1 + k0, K);
#pragma unroll
    for (int c = 0; c < 4; ++c) {
      v16u b = load_bT_frag(W + (size_t)(n0 + c * 16) * K + k0, K);
      acc[0][c] = wmma_bf16(a0, b, acc[0][c]);
      acc[1][c] = wmma_bf16(a1, b, acc[1][c]);
    }
  }
  int hi = (lane >> 4) * 8;   // C layout: lanes>=16 hold rows 8..15
  int cb = lane & 15;
#pragma unroll
  for (int i = 0; i < 2; ++i) {
#pragma unroll
    for (int c = 0; c < 4; ++c) {
#pragma unroll
      for (int r = 0; r < 8; ++r) {
        size_t row = (size_t)(m0 + i * 16 + r + hi);
        int col = n0 + c * 16 + cb;
        float v = acc[i][c][r];
        if (FINAL) Yf[row * N + col] = v + bias[col];
        else       Ybf[row * N + col] = f2bf(v);
      }
    }
  }
}

// ---------------------------------------------------------------------------
// Flash attention: one wave per (b, head, 16-row q tile). Q/K/V are bf16 in
// [B*S, HH] with head n occupying columns n*64..n*64+63.
__global__ __launch_bounds__(32) void attn_kernel(
    const unsigned short* __restrict__ Qb, const unsigned short* __restrict__ Kb,
    const unsigned short* __restrict__ Vb, unsigned short* __restrict__ Hb) {
  __shared__ unsigned short ldsP[16 * 32];  // P tile (bf16), C->A layout bridge
  __shared__ unsigned short ldsV[32 * 64];  // V chunk stage

  int lane = threadIdx.x & 31;
  int q0 = blockIdx.x * 16;
  int n  = blockIdx.y;
  int b  = blockIdx.z;
  int hi = lane >> 4;
  int cb = lane & 15;

  const unsigned short* qbase = Qb + ((size_t)(b * SS + q0)) * HH + n * DHEAD;
  v16u a0 = load_a_frag(qbase, HH);        // d = 0..31
  v16u a1 = load_a_frag(qbase + 32, HH);   // d = 32..63

  float mrow[8], lrow[8];
  v8f acc[4] = {};
#pragma unroll
  for (int r = 0; r < 8; ++r) { mrow[r] = -3.0e38f; lrow[r] = 0.0f; }
  const float KS = 0.125f;                 // 1/sqrt(64)
  const float L2E = 1.44269504088896340736f;

  for (int sk0 = 0; sk0 < SS; sk0 += 32) {
    // ---- stage V chunk (32 x 64) into LDS first (async if available) -------
    const unsigned short* vb = Vb + ((size_t)(b * SS + sk0)) * HH + n * DHEAD;
#pragma unroll
    for (int i = 0; i < 8; ++i) {
      int off = (lane + 32 * i) * 8;  // element offset in the 32x64 tile
      int row = off >> 6;
      int col = off & 63;
#if HAVE_ASYNC_COPY
      __builtin_amdgcn_global_load_async_to_lds_b128(
          (AS_GLOBAL v4i_n*)(vb + (size_t)row * HH + col),
          (AS_LDS v4i_n*)(ldsV + off), 0, 0);
#else
      *(v4u*)(ldsV + off) = *(const v4u*)(vb + (size_t)row * HH + col);
#endif
    }

    // ---- scores: two 16x16 C tiles over K=64 (2 WMMA each) -----------------
    const unsigned short* kb0 = Kb + ((size_t)(b * SS + sk0)) * HH + n * DHEAD;
    const unsigned short* kb1 = kb0 + (size_t)16 * HH;
    v8f c0 = {}, c1 = {};
    c0 = wmma_bf16(a0, load_bT_frag(kb0, HH), c0);
    c0 = wmma_bf16(a1, load_bT_frag(kb0 + 32, HH), c0);
    c1 = wmma_bf16(a0, load_bT_frag(kb1, HH), c1);
    c1 = wmma_bf16(a1, load_bT_frag(kb1 + 32, HH), c1);

    // ---- online softmax (rows live per half-wave; 16-lane reductions) ------
#pragma unroll
    for (int r = 0; r < 8; ++r) {
      float s0 = c0[r] * KS;
      float s1 = c1[r] * KS;
      float tm = fmaxf(s0, s1);
#pragma unroll
      for (int off = 1; off < 16; off <<= 1)
        tm = fmaxf(tm, __shfl_xor(tm, off, 32));
      float nm = fmaxf(mrow[r], tm);
      float corr = __builtin_amdgcn_exp2f((mrow[r] - nm) * L2E);
      float p0 = __builtin_amdgcn_exp2f((s0 - nm) * L2E);
      float p1 = __builtin_amdgcn_exp2f((s1 - nm) * L2E);
      float ps = p0 + p1;
#pragma unroll
      for (int off = 1; off < 16; off <<= 1)
        ps += __shfl_xor(ps, off, 32);
      mrow[r] = nm;
      lrow[r] = lrow[r] * corr + ps;
#pragma unroll
      for (int j = 0; j < 4; ++j) acc[j][r] = acc[j][r] * corr;
      int prow = r + hi * 8;
      ldsP[prow * 32 + cb]      = f2bf(p0);
      ldsP[prow * 32 + 16 + cb] = f2bf(p1);
    }

#if HAVE_ASYNC_COPY
    __builtin_amdgcn_s_wait_asynccnt(0);
#endif
    __syncthreads();

    // ---- PV: P (16x32) as A-frag, V as 4 B-frags -> 4 WMMA -----------------
    v16u pA = load_a_frag(ldsP, 32);
#pragma unroll
    for (int j = 0; j < 4; ++j) {
      v16u bv = load_b_frag_lds(ldsV + j * 16, 64);
      acc[j] = wmma_bf16(pA, bv, acc[j]);
    }
    __syncthreads();
  }

  // ---- epilogue: heads[b,q,n*64+d] = acc/l (bf16) --------------------------
  unsigned short* ob = Hb + ((size_t)(b * SS + q0)) * HH + n * DHEAD;
#pragma unroll
  for (int j = 0; j < 4; ++j) {
#pragma unroll
    for (int r = 0; r < 8; ++r) {
      int row = r + hi * 8;
      int col = j * 16 + cb;
      ob[(size_t)row * HH + col] = f2bf(acc[j][r] / lrow[r]);
    }
  }
}

// ---------------------------------------------------------------------------
extern "C" void kernel_launch(void* const* d_in, const int* in_sizes, int n_in,
                              void* d_out, int out_size, void* d_ws, size_t ws_size,
                              hipStream_t stream) {
  (void)in_sizes; (void)n_in; (void)out_size; (void)ws_size;
  const float* dec  = (const float*)d_in[0];
  const float* enc  = (const float*)d_in[1];
  const float* wq   = (const float*)d_in[2];
  const float* wk   = (const float*)d_in[3];
  const float* wv   = (const float*)d_in[4];
  const float* wo   = (const float*)d_in[5];
  const float* bout = (const float*)d_in[6];
  float* out = (float*)d_out;

  const size_t nX = (size_t)BB * SS * HH;  // 8,388,608
  const size_t nW = (size_t)HH * HH;       // 1,048,576

  char* p = (char*)d_ws;
  size_t o = 0;
  auto take = [&](size_t elems) {
    unsigned short* r = (unsigned short*)(p + o);
    o += ((elems * 2) + 255) & ~(size_t)255;
    return r;
  };
  unsigned short* Xd  = take(nX);
  unsigned short* Xe  = take(nX);
  unsigned short* Wqb = take(nW);
  unsigned short* Wkb = take(nW);
  unsigned short* Wvb = take(nW);
  unsigned short* Wob = take(nW);
  unsigned short* Qb  = take(nX);
  unsigned short* Kb  = take(nX);
  unsigned short* Vb  = take(nX);
  unsigned short* Hb  = take(nX);

  // 1) f32 -> bf16 conversions
  cvt_f32_bf16<<<(int)((nX + 255) / 256), 256, 0, stream>>>(dec, Xd, (int)nX);
  cvt_f32_bf16<<<(int)((nX + 255) / 256), 256, 0, stream>>>(enc, Xe, (int)nX);
  cvt_f32_bf16<<<(int)((nW + 255) / 256), 256, 0, stream>>>(wq, Wqb, (int)nW);
  cvt_f32_bf16<<<(int)((nW + 255) / 256), 256, 0, stream>>>(wk, Wkb, (int)nW);
  cvt_f32_bf16<<<(int)((nW + 255) / 256), 256, 0, stream>>>(wv, Wvb, (int)nW);
  cvt_f32_bf16<<<(int)((nW + 255) / 256), 256, 0, stream>>>(wo, Wob, (int)nW);

  // 2) QKV projections: [8192,1024] x [1024,1024]^T
  dim3 gg(BB * SS / 256, HH / 64);
  gemm_xwT<false><<<gg, 256, 0, stream>>>(Xd, Wqb, Qb, nullptr, nullptr,
                                          BB * SS, HH, HH);
  gemm_xwT<false><<<gg, 256, 0, stream>>>(Xe, Wkb, Kb, nullptr, nullptr,
                                          BB * SS, HH, HH);
  gemm_xwT<false><<<gg, 256, 0, stream>>>(Xe, Wvb, Vb, nullptr, nullptr,
                                          BB * SS, HH, HH);

  // 3) flash attention: one wave per (qtile, head, batch)
  dim3 ga(SS / 16, NHEAD, BB);
  attn_kernel<<<ga, 32, 0, stream>>>(Qb, Kb, Vb, Hb);

  // 4) output projection + bias -> f32
  gemm_xwT<true><<<gg, 256, 0, stream>>>(Hb, Wob, nullptr, out, bout,
                                         BB * SS, HH, HH);
}
